// GPT_89455578841311
// MI455X (gfx1250) — compile-verified, matrix operand order checked by reading, not run on hardware
//
#include <hip/hip_runtime.h>
#include <hip/hip_bf16.h>
#include <math.h>
#include <stdint.h>

// ---------------------------------------------------------------------------
// GPT-2-ish forward (6 layers, B=4, T=1024, D=1024, H=16, V=50257) on gfx1250.
// All contractions on v_wmma_f32_16x16x32_bf16. GEMM operands staged into LDS
// with gfx1250 async copies (global_load_async_to_lds_b128 / ASYNCcnt /
// s_wait_asynccnt), double-buffered, fragments re-read as ds_load_b128.
// ---------------------------------------------------------------------------

#define GPT_L   6
#define GPT_B   4
#define GPT_T   1024
#define GPT_D   1024
#define GPT_H   16
#define GPT_V   50257
#define GPT_DH  64
#define GPT_DFF 4096
#define GPT_BT  (GPT_B * GPT_T)
#define ATTN_SCALE 0.125f   // 1/sqrt(64)

typedef __attribute__((ext_vector_type(16))) __bf16 v16bf_t;
typedef __attribute__((ext_vector_type(8)))  __bf16 v8bf_t;
typedef __attribute__((ext_vector_type(8)))  float  v8f_t;

// 16-bit A/B fragment (16x32) for one lane: elements 0..7  = bf16 at
// k0 + half*8 .. +7, elements 8..15 = bf16 at k0 + 16 + half*8 .. +7.
// (CDNA5 ISA §7.12.2 16-bit A-matrix layout, row/col = lane&15)
__device__ __forceinline__ v16bf_t load_frag_row(const __bf16* __restrict__ rowptr,
                                                 int k0, int half) {
    v8bf_t lo = *(const v8bf_t*)(rowptr + k0 + half * 8);
    v8bf_t hi = *(const v8bf_t*)(rowptr + k0 + 16 + half * 8);
    v16bf_t r;
#pragma unroll
    for (int i = 0; i < 8; ++i) { r[i] = lo[i]; r[8 + i] = hi[i]; }
    return r;
}

__device__ __forceinline__ v8f_t wmma_bf16(v16bf_t a, v16bf_t b, v8f_t c) {
    return __builtin_amdgcn_wmma_f32_16x16x32_bf16(false, a, false, b,
                                                   (short)0, c, false, false);
}

// gfx1250 async copy: 16 bytes global -> LDS, tracked by ASYNCcnt.
// LDS address = low 32 bits of the generic pointer (ISA §10.2 aperture rule).
__device__ __forceinline__ void async_ld_b128(void* lds_ptr, const void* gptr) {
    uint32_t loff = (uint32_t)(uintptr_t)lds_ptr;
    unsigned long long ga = (unsigned long long)(uintptr_t)gptr;
    asm volatile("global_load_async_to_lds_b128 %0, %1, off"
                 :: "v"(loff), "v"(ga) : "memory");
}

// ---------------------------------------------------------------------------
// Weight transpose + convert: W[K,N] f32 (row-major) -> WT[N,K] bf16.
// ---------------------------------------------------------------------------
__global__ void k_wtrans(const float* __restrict__ W, __bf16* __restrict__ WT,
                         int K, int N) {
    __shared__ float tile[32][33];
    const int n0 = blockIdx.x * 32;
    const int k0 = blockIdx.y * 32;
    const int tx = threadIdx.x;        // 0..31
    const int ty = threadIdx.y;        // 0..7
#pragma unroll
    for (int i = 0; i < 4; ++i) {
        int kk = k0 + ty + i * 8;
        int nn = n0 + tx;
        tile[ty + i * 8][tx] = (nn < N) ? W[(long)kk * N + nn] : 0.0f;
    }
    __syncthreads();
#pragma unroll
    for (int i = 0; i < 4; ++i) {
        int nn = n0 + ty + i * 8;
        int kk = k0 + tx;
        if (nn < N) WT[(long)nn * K + kk] = (__bf16)tile[tx][ty + i * 8];
    }
}

// ---------------------------------------------------------------------------
// V transpose (bf16): v[B*T, D] -> vt[B*H*DH, T]
// ---------------------------------------------------------------------------
__global__ void k_vtrans(const __bf16* __restrict__ v, __bf16* __restrict__ vt) {
    __shared__ __bf16 tile[32][33];
    const int bh = blockIdx.z;
    const int d0 = blockIdx.y * 32;
    const int t0 = blockIdx.x * 32;
    const int b  = bh / GPT_H;
    const int h  = bh % GPT_H;
    const int tx = threadIdx.x, ty = threadIdx.y;
#pragma unroll
    for (int i = 0; i < 4; ++i) {
        int t = t0 + ty + i * 8;
        tile[ty + i * 8][tx] = v[((long)b * GPT_T + t) * GPT_D + h * GPT_DH + d0 + tx];
    }
    __syncthreads();
#pragma unroll
    for (int i = 0; i < 4; ++i) {
        int d = d0 + ty + i * 8;
        vt[((long)bh * GPT_DH + d) * GPT_T + t0 + tx] = tile[tx][ty + i * 8];
    }
}

// ---------------------------------------------------------------------------
// Embedding: x[b,t,:] = tok_emb[idx[b,t],:] + pos_emb[t,:]   (fp32 residual)
// ---------------------------------------------------------------------------
__global__ void k_embed(const int* __restrict__ idx, const float* __restrict__ tok,
                        const float* __restrict__ pos, float* __restrict__ x) {
    long bt = blockIdx.x;
    int  t  = (int)(bt % GPT_T);
    long tk = idx[bt];
    const float* te = tok + tk * GPT_D;
    const float* pe = pos + (long)t * GPT_D;
    float* xr = x + bt * GPT_D;
    for (int i = threadIdx.x; i < GPT_D; i += blockDim.x) xr[i] = te[i] + pe[i];
}

// ---------------------------------------------------------------------------
// LayerNorm over D=1024, fp32 in, bf16 out. One 256-thread block per row.
// ---------------------------------------------------------------------------
__global__ void k_layernorm(const float* __restrict__ x, const float* __restrict__ g,
                            const float* __restrict__ b, __bf16* __restrict__ out) {
    long row = blockIdx.x;
    const float* xr = x + row * GPT_D;
    __bf16* outr = out + row * GPT_D;

    float s = 0.f, s2 = 0.f;
    for (int i = threadIdx.x; i < GPT_D; i += blockDim.x) {
        float v = xr[i]; s += v; s2 += v * v;
    }
    __shared__ float red[64];
#pragma unroll
    for (int off = 16; off; off >>= 1) {
        s  += __shfl_down(s, off);
        s2 += __shfl_down(s2, off);
    }
    int wid = threadIdx.x >> 5, ln = threadIdx.x & 31;
    if (ln == 0) { red[wid] = s; red[wid + 32] = s2; }
    __syncthreads();
    if (threadIdx.x == 0) {
        float ts = 0.f, ts2 = 0.f;
        for (int w = 0; w < (int)(blockDim.x >> 5); ++w) { ts += red[w]; ts2 += red[w + 32]; }
        red[0] = ts; red[32] = ts2;
    }
    __syncthreads();
    float mu   = red[0] * (1.0f / GPT_D);
    float var  = red[32] * (1.0f / GPT_D) - mu * mu;
    float rstd = rsqrtf(var + 1e-5f);
    for (int i = threadIdx.x; i < GPT_D; i += blockDim.x)
        outr[i] = (__bf16)((xr[i] - mu) * rstd * g[i] + b[i]);
}

// ---------------------------------------------------------------------------
// WMMA GEMM with async-LDS double-buffered staging.
//   A:  bf16 [M,K] row-major;  BT: bf16 [N,K] row-major (weights^T)
//   Block (256 thr, 8 waves) computes a 128x128 C tile:
//     wave w: mgroup = w&1 (64 rows), ngroup = w>>1 (32 cols = 2 n-tiles).
//   Per K-step of 32: every thread issues exactly 4
//   global_load_async_to_lds_b128 chunks (A panel 128x32 + B panel 128x32),
//   so the steady-state wait is one uniform s_wait_asynccnt 4. Last
//   iteration peeled -> branch-free hot loop. 8 WMMAs / 12 ds_load_b128.
// ---------------------------------------------------------------------------
#define AST 40   // LDS panel row stride (elements): 80B, 16B-aligned, conflict-free

__global__ void __launch_bounds__(256)
k_gemm(const __bf16* __restrict__ A, const __bf16* __restrict__ BT,
       float* __restrict__ Cf, __bf16* __restrict__ Cb,
       const float* __restrict__ bias, const float* __restrict__ resid,
       int M, int N, int K, int relu) {
    __shared__ __align__(16) __bf16 As[2][128 * AST];
    __shared__ __align__(16) __bf16 Bs[2][128 * AST];

    const int tid  = threadIdx.x;
    const int lane = tid & 31;
    const int wave = tid >> 5;
    const int half = lane >> 4, c = lane & 15;

    const int ntn128 = (N + 127) >> 7;
    const int bm = blockIdx.x / ntn128;
    const int bn = blockIdx.x % ntn128;
    const int m0b = bm << 7;
    const int n0b = bn << 7;

    const int mg = wave & 1;          // 0..1 : which 64 rows
    const int ng = wave >> 1;         // 0..3 : which 32 cols

    // symmetric staging: thread -> rows r and r+64, chunk sub (4x16B per row)
    const int r   = tid >> 2;         // 0..63
    const int sub = tid & 3;          // 0..3
    const __bf16* agp0 = A + (long)(m0b + r) * K + sub * 8;
    const __bf16* agp1 = agp0 + (long)64 * K;
    const int nbr0 = min(n0b + r, N - 1);          // ragged-N clamp
    const int nbr1 = min(n0b + 64 + r, N - 1);
    const __bf16* bgp0 = BT + (long)nbr0 * K + sub * 8;
    const __bf16* bgp1 = BT + (long)nbr1 * K + sub * 8;
    const int la0 = r * AST + sub * 8;
    const int la1 = (64 + r) * AST + sub * 8;

    v8f_t acc[2][4] = {};
    const int nIter = K >> 5;

    // prologue: stage K-step 0 into buffer 0
    async_ld_b128(&As[0][la0], agp0);
    async_ld_b128(&As[0][la1], agp1);
    async_ld_b128(&Bs[0][la0], bgp0);
    async_ld_b128(&Bs[0][la1], bgp1);

    for (int it = 0; it < nIter - 1; ++it) {
        const int buf = it & 1;
        const long ko = (long)(it + 1) << 5;
        async_ld_b128(&As[buf ^ 1][la0], agp0 + ko);
        async_ld_b128(&As[buf ^ 1][la1], agp1 + ko);
        async_ld_b128(&Bs[buf ^ 1][la0], bgp0 + ko);
        async_ld_b128(&Bs[buf ^ 1][la1], bgp1 + ko);
        // ASYNCcnt retires in order: allow the 4 just-issued next-stage ops.
        asm volatile("s_wait_asynccnt 4" ::: "memory");
        __syncthreads();

        v16bf_t b0 = load_frag_row(&Bs[buf][(ng * 32 + c) * AST], 0, half);
        v16bf_t b1 = load_frag_row(&Bs[buf][(ng * 32 + 16 + c) * AST], 0, half);
#pragma unroll
        for (int mt = 0; mt < 4; ++mt) {
            v16bf_t a = load_frag_row(&As[buf][(mg * 64 + mt * 16 + c) * AST], 0, half);
            acc[0][mt] = wmma_bf16(a, b0, acc[0][mt]);
            acc[1][mt] = wmma_bf16(a, b1, acc[1][mt]);
        }
        __syncthreads();   // protect buffers before next overwrite
    }

    // peeled last iteration (no staging, wait for everything)
    {
        const int buf = (nIter - 1) & 1;
        asm volatile("s_wait_asynccnt 0" ::: "memory");
        __syncthreads();
        v16bf_t b0 = load_frag_row(&Bs[buf][(ng * 32 + c) * AST], 0, half);
        v16bf_t b1 = load_frag_row(&Bs[buf][(ng * 32 + 16 + c) * AST], 0, half);
#pragma unroll
        for (int mt = 0; mt < 4; ++mt) {
            v16bf_t a = load_frag_row(&As[buf][(mg * 64 + mt * 16 + c) * AST], 0, half);
            acc[0][mt] = wmma_bf16(a, b0, acc[0][mt]);
            acc[1][mt] = wmma_bf16(a, b1, acc[1][mt]);
        }
    }

    // epilogue
#pragma unroll
    for (int nt = 0; nt < 2; ++nt) {
        const int n = n0b + ng * 32 + nt * 16 + c;
        if (n < N) {
            const float bn = bias ? bias[n] : 0.0f;
#pragma unroll
            for (int mt = 0; mt < 4; ++mt) {
#pragma unroll
                for (int j = 0; j < 8; ++j) {
                    int m = m0b + mg * 64 + mt * 16 + half * 8 + j;
                    float v = acc[nt][mt][j] + bn;
                    if (relu) v = fmaxf(v, 0.0f);
                    if (resid) v += resid[(long)m * N + n];
                    if (Cf) Cf[(long)m * N + n] = v;
                    if (Cb) Cb[(long)m * N + n] = (__bf16)v;
                }
            }
        }
    }
}

// ---------------------------------------------------------------------------
// Fused causal attention (flash-style), one wave per (b, h, 16-query tile).
// q/k bf16 [B*T, D]; vt bf16 [B*H*DH, T]; out bf16 [B*T, D].
// ---------------------------------------------------------------------------
__global__ void __launch_bounds__(32)
k_attn(const __bf16* __restrict__ Q, const __bf16* __restrict__ Kx,
       const __bf16* __restrict__ VT, __bf16* __restrict__ O) {
    const int lane = threadIdx.x & 31;
    const int nqt  = GPT_T >> 4;
    const int qt   = blockIdx.x % nqt;
    const int bh   = blockIdx.x / nqt;
    const int h    = bh % GPT_H;
    const int b    = bh / GPT_H;
    const int q0   = qt << 4;
    const int half = lane >> 4, c = lane & 15;

    const __bf16* Qb  = Q  + ((long)b * GPT_T) * GPT_D + h * GPT_DH;
    const __bf16* Kb  = Kx + ((long)b * GPT_T) * GPT_D + h * GPT_DH;
    const __bf16* VTb = VT + (long)bh * GPT_DH * GPT_T;

    v16bf_t qa0 = load_frag_row(Qb + (long)(q0 + c) * GPT_D, 0,  half);
    v16bf_t qa1 = load_frag_row(Qb + (long)(q0 + c) * GPT_D, 32, half);

    v8f_t o0 = {}, o1 = {}, o2 = {}, o3 = {};
    float mrow[8], lrow[8];
#pragma unroll
    for (int j = 0; j < 8; ++j) { mrow[j] = -1e30f; lrow[j] = 0.0f; }

    __shared__ __align__(16) __bf16 Pl[16][32];

    const int nkc = (q0 + 16 + 31) >> 5;
    for (int kc = 0; kc < nkc; ++kc) {
        const int kbase = kc << 5;

        v8f_t s0 = {}, s1 = {};
        {
            const __bf16* kr0 = Kb + (long)(kbase + c) * GPT_D;
            const __bf16* kr1 = Kb + (long)(kbase + 16 + c) * GPT_D;
            s0 = wmma_bf16(qa0, load_frag_row(kr0, 0,  half), s0);
            s0 = wmma_bf16(qa1, load_frag_row(kr0, 32, half), s0);
            s1 = wmma_bf16(qa0, load_frag_row(kr1, 0,  half), s1);
            s1 = wmma_bf16(qa1, load_frag_row(kr1, 32, half), s1);
        }

#pragma unroll
        for (int j = 0; j < 8; ++j) {
            const int qrow = q0 + half * 8 + j;
            float v0 = s0[j] * ATTN_SCALE;
            float v1 = s1[j] * ATTN_SCALE;
            if (kbase + c      > qrow) v0 = -1e30f;
            if (kbase + 16 + c > qrow) v1 = -1e30f;

            float mx = fmaxf(v0, v1);
#pragma unroll
            for (int off = 1; off < 16; off <<= 1) mx = fmaxf(mx, __shfl_xor(mx, off));

            const float mnew  = fmaxf(mrow[j], mx);
            const float alpha = __expf(mrow[j] - mnew);
            const float p0 = __expf(v0 - mnew);
            const float p1 = __expf(v1 - mnew);
            float rs = p0 + p1;
#pragma unroll
            for (int off = 1; off < 16; off <<= 1) rs += __shfl_xor(rs, off);

            lrow[j] = lrow[j] * alpha + rs;
            mrow[j] = mnew;
            o0[j] *= alpha; o1[j] *= alpha; o2[j] *= alpha; o3[j] *= alpha;

            Pl[half * 8 + j][c]      = (__bf16)p0;
            Pl[half * 8 + j][16 + c] = (__bf16)p1;
        }
        __syncthreads();

        v16bf_t pa;
        {
            v8bf_t lo = *(const v8bf_t*)&Pl[c][half * 8];
            v8bf_t hi = *(const v8bf_t*)&Pl[c][16 + half * 8];
#pragma unroll
            for (int i = 0; i < 8; ++i) { pa[i] = lo[i]; pa[8 + i] = hi[i]; }
        }
        __syncthreads();

        o0 = wmma_bf16(pa, load_frag_row(VTb + (long)(c)      * GPT_T, kbase, half), o0);
        o1 = wmma_bf16(pa, load_frag_row(VTb + (long)(16 + c) * GPT_T, kbase, half), o1);
        o2 = wmma_bf16(pa, load_frag_row(VTb + (long)(32 + c) * GPT_T, kbase, half), o2);
        o3 = wmma_bf16(pa, load_frag_row(VTb + (long)(48 + c) * GPT_T, kbase, half), o3);
    }

#pragma unroll
    for (int j = 0; j < 8; ++j) {
        const float inv = 1.0f / lrow[j];
        const int m = q0 + half * 8 + j;
        __bf16* orow = O + ((long)b * GPT_T + m) * GPT_D + h * GPT_DH;
        orow[c]      = (__bf16)(o0[j] * inv);
        orow[16 + c] = (__bf16)(o1[j] * inv);
        orow[32 + c] = (__bf16)(o2[j] * inv);
        orow[48 + c] = (__bf16)(o3[j] * inv);
    }
}

// ---------------------------------------------------------------------------
// Loss: per-row log-softmax NLL over V, accumulated (mean) into *loss.
// ---------------------------------------------------------------------------
__global__ void k_zero1(float* p) { *p = 0.0f; }

__global__ void k_loss(const float* __restrict__ logits, const int* __restrict__ targets,
                       float* __restrict__ loss, float invBT) {
    long row = blockIdx.x;
    const float* lr = logits + row * (long)GPT_V;
    __shared__ float red[32];
    int wid = threadIdx.x >> 5, ln = threadIdx.x & 31;

    float mx = -1e30f;
    for (int i = threadIdx.x; i < GPT_V; i += blockDim.x) mx = fmaxf(mx, lr[i]);
#pragma unroll
    for (int off = 16; off; off >>= 1) mx = fmaxf(mx, __shfl_down(mx, off));
    if (ln == 0) red[wid] = mx;
    __syncthreads();
    if (threadIdx.x == 0) {
        float m = red[0];
        for (int w = 1; w < (int)(blockDim.x >> 5); ++w) m = fmaxf(m, red[w]);
        red[0] = m;
    }
    __syncthreads();
    mx = red[0];
    __syncthreads();

    float s = 0.0f;
    for (int i = threadIdx.x; i < GPT_V; i += blockDim.x) s += __expf(lr[i] - mx);
#pragma unroll
    for (int off = 16; off; off >>= 1) s += __shfl_down(s, off);
    if (ln == 0) red[wid] = s;
    __syncthreads();
    if (threadIdx.x == 0) {
        float tot = 0.0f;
        for (int w = 0; w < (int)(blockDim.x >> 5); ++w) tot += red[w];
        int tg = targets[row];
        float lp = lr[tg] - mx - __logf(tot);
        atomicAdd(loss, -lp * invBT);
    }
}

// ---------------------------------------------------------------------------
// Host orchestration
// ---------------------------------------------------------------------------
static inline void launch_wtrans(const float* W, __bf16* WT, int K, int N,
                                 hipStream_t s) {
    dim3 grid((N + 31) / 32, K / 32);
    k_wtrans<<<grid, dim3(32, 8), 0, s>>>(W, WT, K, N);
}

static inline void launch_gemm(const __bf16* A, const __bf16* BT,
                               float* Cf, __bf16* Cb,
                               const float* bias, const float* resid,
                               int M, int N, int K, int relu, hipStream_t s) {
    int ntn128 = (N + 127) / 128;
    int blocks = (M / 128) * ntn128;
    k_gemm<<<blocks, 256, 0, s>>>(A, BT, Cf, Cb, bias, resid, M, N, K, relu);
}

extern "C" void kernel_launch(void* const* d_in, const int* in_sizes, int n_in,
                              void* d_out, int out_size, void* d_ws, size_t ws_size,
                              hipStream_t stream) {
    (void)in_sizes; (void)n_in; (void)out_size; (void)ws_size;

    const int*   idx   = (const int*)  d_in[0];
    const int*   tgt   = (const int*)  d_in[1];
    const float* tok   = (const float*)d_in[2];
    const float* pos   = (const float*)d_in[3];
    const float* Wq    = (const float*)d_in[4];
    const float* Wk    = (const float*)d_in[5];
    const float* Wv    = (const float*)d_in[6];
    const float* Wo    = (const float*)d_in[7];
    const float* bo    = (const float*)d_in[8];
    const float* ln1g  = (const float*)d_in[9];
    const float* ln1b  = (const float*)d_in[10];
    const float* ln2g  = (const float*)d_in[11];
    const float* ln2b  = (const float*)d_in[12];
    const float* W1    = (const float*)d_in[13];
    const float* b1    = (const float*)d_in[14];
    const float* W2    = (const float*)d_in[15];
    const float* b2    = (const float*)d_in[16];
    const float* lng   = (const float*)d_in[17];
    const float* lnb   = (const float*)d_in[18];
    const float* Whead = (const float*)d_in[19];
    const float* bhead = (const float*)d_in[20];

    float* logits = (float*)d_out;
    float* loss   = logits + (size_t)GPT_BT * GPT_V;

    // workspace: wt | x(f32) | h | q | k | v | ao | vt | ff
    const size_t SD = (size_t)GPT_BT * GPT_D;
    char* p = (char*)d_ws;
    __bf16* wt = (__bf16*)p;  p += sizeof(__bf16) * (size_t)GPT_V * GPT_D;
    float*  x  = (float*)p;   p += sizeof(float)  * SD;
    __bf16* h  = (__bf16*)p;  p += sizeof(__bf16) * SD;
    __bf16* q  = (__bf16*)p;  p += sizeof(__bf16) * SD;
    __bf16* k  = (__bf16*)p;  p += sizeof(__bf16) * SD;
    __bf16* v  = (__bf16*)p;  p += sizeof(__bf16) * SD;
    __bf16* ao = (__bf16*)p;  p += sizeof(__bf16) * SD;
    __bf16* vt = (__bf16*)p;  p += sizeof(__bf16) * SD;
    __bf16* ff = (__bf16*)p;  p += sizeof(__bf16) * (size_t)GPT_BT * GPT_DFF;

    k_embed<<<GPT_BT, 256, 0, stream>>>(idx, tok, pos, x);

    for (int l = 0; l < GPT_L; ++l) {
        const size_t wofs = (size_t)l * GPT_D * GPT_D;
        const size_t fofs = (size_t)l * GPT_D * GPT_DFF;

        k_layernorm<<<GPT_BT, 256, 0, stream>>>(x, ln1g + (size_t)l * GPT_D,
                                                ln1b + (size_t)l * GPT_D, h);

        launch_wtrans(Wq + wofs, wt, GPT_D, GPT_D, stream);
        launch_gemm(h, wt, nullptr, q, nullptr, nullptr, GPT_BT, GPT_D, GPT_D, 0, stream);
        launch_wtrans(Wk + wofs, wt, GPT_D, GPT_D, stream);
        launch_gemm(h, wt, nullptr, k, nullptr, nullptr, GPT_BT, GPT_D, GPT_D, 0, stream);
        launch_wtrans(Wv + wofs, wt, GPT_D, GPT_D, stream);
        launch_gemm(h, wt, nullptr, v, nullptr, nullptr, GPT_BT, GPT_D, GPT_D, 0, stream);

        k_vtrans<<<dim3(GPT_T / 32, GPT_DH / 32, GPT_B * GPT_H), dim3(32, 8), 0, stream>>>(v, vt);
        k_attn<<<GPT_B * GPT_H * (GPT_T / 16), 32, 0, stream>>>(q, k, vt, ao);

        launch_wtrans(Wo + wofs, wt, GPT_D, GPT_D, stream);
        launch_gemm(ao, wt, x, nullptr, bo + (size_t)l * GPT_D, x,
                    GPT_BT, GPT_D, GPT_D, 0, stream);

        k_layernorm<<<GPT_BT, 256, 0, stream>>>(x, ln2g + (size_t)l * GPT_D,
                                                ln2b + (size_t)l * GPT_D, h);

        launch_wtrans(W1 + fofs, wt, GPT_D, GPT_DFF, stream);
        launch_gemm(h, wt, nullptr, ff, b1 + (size_t)l * GPT_DFF, nullptr,
                    GPT_BT, GPT_DFF, GPT_D, 1, stream);
        launch_wtrans(W2 + fofs, wt, GPT_DFF, GPT_D, stream);
        launch_gemm(ff, wt, x, nullptr, b2 + (size_t)l * GPT_D, x,
                    GPT_BT, GPT_D, GPT_DFF, 0, stream);
    }

    k_layernorm<<<GPT_BT, 256, 0, stream>>>(x, lng, lnb, h);
    launch_wtrans(Whead, wt, GPT_D, GPT_V, stream);
    launch_gemm(h, wt, logits, nullptr, bhead, nullptr,
                GPT_BT, GPT_V, GPT_D, 0, stream);

    k_zero1<<<1, 1, 0, stream>>>(loss);
    k_loss<<<GPT_BT, 256, 0, stream>>>(logits, tgt, loss, 1.0f / (float)GPT_BT);
}
